// FaceEncoder2_76416058130784
// MI455X (gfx1250) — matrix operand order repaired and checked by reading, not compiled, not run on hardware
//
#include <hip/hip_runtime.h>

#define NN    14976
#define EE    89600
#define E2    104576      /* EE + NN self loops */
#define HID   256
#define OUTC  512
#define HEADS 4
#define BGR   32
#define NPG   468
#define NEGSL 0.2f
#define EPSF  1e-16f

typedef _Float16 f16;
typedef __attribute__((ext_vector_type(16))) _Float16 v16h;
typedef __attribute__((ext_vector_type(8)))  float    v8f;
typedef long long ll;

// ---- ordered-uint encoding for float atomic max ----
__device__ __forceinline__ unsigned f2ord(float f) {
  unsigned u = __float_as_uint(f);
  return (u & 0x80000000u) ? ~u : (u | 0x80000000u);
}
__device__ __forceinline__ float ord2f(unsigned u) {
  return (u & 0x80000000u) ? __uint_as_float(u & 0x7FFFFFFFu) : __uint_as_float(~u);
}

// p1[sel*8 + h*2 + i] = sum_c W1[i,h*256+c] * a1(sel)[h,c]      (16 threads)
__global__ void prep_p1(const float* __restrict__ W1, const float* __restrict__ a1s,
                        const float* __restrict__ a1d, float* __restrict__ p1) {
  int t = threadIdx.x; if (t >= 16) return;
  int sel = t >> 3, h = (t & 7) >> 1, i = t & 1;
  const float* a = sel ? a1d : a1s;
  float acc = 0.f;
  for (int c = 0; c < HID; ++c) acc += W1[i * 1024 + h * HID + c] * a[h * HID + c];
  p1[t] = acc;
}

// q2[sel*1024 + h*256 + k] = sum_c W2[k, h*512+c] * a2(sel)[h,c]   (2048 threads)
__global__ void prep_q2(const float* __restrict__ W2, const float* __restrict__ a2s,
                        const float* __restrict__ a2d, float* __restrict__ q2) {
  int idx = blockIdx.x * 256 + threadIdx.x;      // 0..2047
  int sel = idx >> 10, r = idx & 1023, h = r >> 8, k = r & 255;
  const float* a = sel ? a2d : a2s;
  float acc = 0.f;
  for (int c = 0; c < OUTC; ++c) acc += W2[(size_t)k * 2048 + h * OUTC + c] * a[h * OUTC + c];
  q2[idx] = acc;
}

// W2T f16 [2048, 256]: W2T[j,k] = W2[k,j]
__global__ void prep_w2t(const float* __restrict__ W2, f16* __restrict__ w2t) {
  int idx = blockIdx.x * 256 + threadIdx.x;      // 524288
  int j = idx >> 8, k = idx & 255;
  w2t[(size_t)j * 256 + k] = (f16)W2[(size_t)k * 2048 + j];
}

// per-node layer-1 logits
__global__ void node1_alpha(const float* __restrict__ x, const float* __restrict__ p1,
                            float* __restrict__ as1, float* __restrict__ ad1) {
  int n = blockIdx.x * 256 + threadIdx.x; if (n >= NN) return;
  float x0 = x[n * 2], x1 = x[n * 2 + 1];
#pragma unroll
  for (int h = 0; h < HEADS; ++h) {
    as1[n * 4 + h] = x0 * p1[h * 2 + 0] + x1 * p1[h * 2 + 1];
    ad1[n * 4 + h] = x0 * p1[8 + h * 2 + 0] + x1 * p1[8 + h * 2 + 1];
  }
}

// per-node layer-2 logits: dot(h1[n,:], q2[sel,h,:])
__global__ void node2_alpha(const float* __restrict__ h1, const float* __restrict__ q2,
                            float* __restrict__ as2, float* __restrict__ ad2) {
  int idx = blockIdx.x * 256 + threadIdx.x;      // NN*8
  int n = idx >> 3, r = idx & 7, sel = r >> 2, h = r & 3;
  const float* q = q2 + sel * 1024 + h * 256;
  const float* hr = h1 + (size_t)n * 256;
  float acc = 0.f;
  for (int k = 0; k < 256; ++k) acc += hr[k] * q[k];
  (sel ? ad2 : as2)[n * 4 + h] = acc;
}

// pass A: leaky-relu logits + atomic segment max (ordered uint)
__global__ void edge_logits_max(const ll* __restrict__ ei, const float* __restrict__ as,
                                const float* __restrict__ ad, float* __restrict__ ev,
                                unsigned* __restrict__ m) {
  int e = blockIdx.x * 256 + threadIdx.x; if (e >= E2) return;
  ll s, d;
  if (e < EE) { s = ei[e]; d = ei[EE + e]; } else { s = d = (ll)(e - EE); }
#pragma unroll
  for (int h = 0; h < HEADS; ++h) {
    float v = as[s * 4 + h] + ad[d * 4 + h];
    v = v > 0.f ? v : NEGSL * v;
    ev[(size_t)e * 4 + h] = v;
    atomicMax(&m[d * 4 + h], f2ord(v));
  }
}

// pass B: exp(e - max), atomic denom sum; ev overwritten with exp value
__global__ void edge_exp_sum(const ll* __restrict__ ei, float* __restrict__ ev,
                             const unsigned* __restrict__ m, float* __restrict__ den) {
  int e = blockIdx.x * 256 + threadIdx.x; if (e >= E2) return;
  ll d = (e < EE) ? ei[EE + e] : (ll)(e - EE);
#pragma unroll
  for (int h = 0; h < HEADS; ++h) {
    float ex = __expf(ev[(size_t)e * 4 + h] - ord2f(m[d * 4 + h]));
    ev[(size_t)e * 4 + h] = ex;
    atomicAdd(&den[d * 4 + h], ex);
  }
}

// layer-1 aggregation: messages recomputed from LDS-resident W1, head-mean folded in
__global__ __launch_bounds__(256) void edge1_agg(const ll* __restrict__ ei,
                                                 const float* __restrict__ x,
                                                 const float* __restrict__ W1,
                                                 const float* __restrict__ ev,
                                                 const float* __restrict__ den,
                                                 float* __restrict__ agg1) {
  __shared__ float sW1[2048];
  for (int i = threadIdx.x; i < 2048; i += 256) sW1[i] = W1[i];
  __syncthreads();
  int e0 = blockIdx.x * 16;
  int c = threadIdx.x;
  for (int s = 0; s < 16; ++s) {
    int e = e0 + s; if (e >= E2) break;
    ll sr, ds;
    if (e < EE) { sr = ei[e]; ds = ei[EE + e]; } else { sr = ds = (ll)(e - EE); }
    float x0 = x[sr * 2], x1 = x[sr * 2 + 1];
    float al[HEADS];
#pragma unroll
    for (int h = 0; h < HEADS; ++h)
      al[h] = 0.25f * ev[(size_t)e * 4 + h] / (den[ds * 4 + h] + EPSF);
    float v = 0.f;
#pragma unroll
    for (int h = 0; h < HEADS; ++h)
      v += al[h] * (x0 * sW1[h * 256 + c] + x1 * sW1[1024 + h * 256 + c]);
    atomicAdd(&agg1[(size_t)ds * 256 + c], v);
  }
}

// bias + relu, write f32 in place + f16 copy for WMMA A
__global__ void post1(float* __restrict__ agg1, const float* __restrict__ b1,
                      f16* __restrict__ h1f16) {
  int idx = blockIdx.x * 256 + threadIdx.x;     // NN*256
  float v = agg1[idx] + b1[idx & 255];
  v = v > 0.f ? v : 0.f;
  agg1[idx] = v;
  h1f16[idx] = (f16)v;
}

// WMMA GEMM: xh2[N,2048] = h1f16[N,256] @ W2 ; B supplied transposed (w2t [2048,256])
// block = 4 waves; wave computes 16(M) x 64(N); K stepped 32 per v_wmma_f32_16x16x32_f16
__global__ __launch_bounds__(128) void gemm2_wmma(const f16* __restrict__ A,
                                                  const f16* __restrict__ Bt,
                                                  f16* __restrict__ D) {
  const int m0   = blockIdx.x * 16;
  const int wave = threadIdx.x >> 5;
  const int lane = threadIdx.x & 31;
  const int hf   = lane >> 4;
  const int lr   = lane & 15;
  const int n0   = blockIdx.y * 256 + wave * 64;

  v8f acc[4] = {};
  const f16* arow = A + (size_t)(m0 + lr) * 256 + 8 * hf;   // two 8-runs: +0, +16
#pragma unroll 1
  for (int kk = 0; kk < 256; kk += 32) {
    v16h af;
    {
      union { v16h v; float4 q[2]; } u;
      u.q[0] = *(const float4*)(arow + kk);        // K = kk + 8*hf + [0,8)
      u.q[1] = *(const float4*)(arow + kk + 16);   // K = kk + 16 + 8*hf + [0,8)
      af = u.v;
    }
#pragma unroll
    for (int j = 0; j < 4; ++j) {
      const f16* brow = Bt + (size_t)(n0 + j * 16 + lr) * 256 + kk + 8 * hf; // two 8-runs: +0, +16
      v16h bf;
      {
        union { v16h v; float4 q[2]; } u;
        u.q[0] = *(const float4*)(brow);
        u.q[1] = *(const float4*)(brow + 16);
        bf = u.v;
      }
      acc[j] = __builtin_amdgcn_wmma_f32_16x16x32_f16(false, af, false, bf,
                                                      (short)0, acc[j], false, false);
    }
  }
#pragma unroll
  for (int j = 0; j < 4; ++j)
#pragma unroll
    for (int r = 0; r < 8; ++r)
      D[(size_t)(m0 + r + 8 * hf) * 2048 + n0 + j * 16 + lr] = (f16)acc[j][r];
}

// layer-2 aggregation: gather xh2 f16 messages, head-mean folded, f32 atomics
__global__ __launch_bounds__(256) void edge2_agg(const ll* __restrict__ ei,
                                                 const f16* __restrict__ xh2,
                                                 const float* __restrict__ ev,
                                                 const float* __restrict__ den,
                                                 float* __restrict__ agg2) {
  int e0 = blockIdx.x * 8;
  for (int s = 0; s < 8; ++s) {
    int e = e0 + s; if (e >= E2) break;
    ll sr, ds;
    if (e < EE) { sr = ei[e]; ds = ei[EE + e]; } else { sr = ds = (ll)(e - EE); }
    float al[HEADS];
#pragma unroll
    for (int h = 0; h < HEADS; ++h)
      al[h] = 0.25f * ev[(size_t)e * 4 + h] / (den[ds * 4 + h] + EPSF);
    const f16* xr = xh2 + (size_t)sr * 2048;
    for (int c = threadIdx.x; c < OUTC; c += 256) {
      float v = 0.f;
#pragma unroll
      for (int h = 0; h < HEADS; ++h) v += al[h] * (float)xr[h * OUTC + c];
      atomicAdd(&agg2[(size_t)ds * OUTC + c], v);
    }
  }
}

__global__ void post2(float* __restrict__ agg2, const float* __restrict__ b2) {
  int idx = blockIdx.x * 256 + threadIdx.x;     // NN*512
  float v = agg2[idx] + b2[idx & 511];
  agg2[idx] = v > 0.f ? v : 0.f;
}

__global__ void gate_k(const float* __restrict__ h2, const float* __restrict__ Wg,
                       const float* __restrict__ bg, float* __restrict__ gate) {
  int n = blockIdx.x * 256 + threadIdx.x; if (n >= NN) return;
  const float* hr = h2 + (size_t)n * OUTC;
  float acc = bg[0];
  for (int c = 0; c < OUTC; ++c) acc += hr[c] * Wg[c];
  gate[n] = acc;
}

// global-attention pooling; one block per graph (nodes contiguous per graph)
__global__ __launch_bounds__(256) void pool_k(const float* __restrict__ h2,
                                              const float* __restrict__ gate,
                                              float* __restrict__ out) {
  __shared__ float red[256];
  __shared__ float attn[NPG];
  __shared__ float sM, sInv;
  int g = blockIdx.x, base = g * NPG, tid = threadIdx.x;

  float lm = -3.0e38f;
  for (int n = tid; n < NPG; n += 256) lm = fmaxf(lm, gate[base + n]);
  red[tid] = lm; __syncthreads();
  for (int st = 128; st > 0; st >>= 1) {
    if (tid < st) red[tid] = fmaxf(red[tid], red[tid + st]);
    __syncthreads();
  }
  if (tid == 0) sM = red[0];
  __syncthreads();

  float ls = 0.f;
  for (int n = tid; n < NPG; n += 256) {
    float ex = __expf(gate[base + n] - sM);
    attn[n] = ex; ls += ex;
  }
  red[tid] = ls; __syncthreads();
  for (int st = 128; st > 0; st >>= 1) {
    if (tid < st) red[tid] += red[tid + st];
    __syncthreads();
  }
  if (tid == 0) sInv = 1.0f / (red[0] + EPSF);
  __syncthreads();

  for (int c = tid; c < OUTC; c += 256) {
    float acc = 0.f;
    for (int n = 0; n < NPG; ++n) acc += attn[n] * h2[(size_t)(base + n) * OUTC + c];
    out[(size_t)g * OUTC + c] = acc * sInv;
  }
}

extern "C" void kernel_launch(void* const* d_in, const int* in_sizes, int n_in,
                              void* d_out, int out_size, void* d_ws, size_t ws_size,
                              hipStream_t stream) {
  const float* x   = (const float*)d_in[0];
  const ll*    ei  = (const ll*)   d_in[1];
  const float* W1  = (const float*)d_in[3];
  const float* a1s = (const float*)d_in[4];
  const float* a1d = (const float*)d_in[5];
  const float* b1  = (const float*)d_in[6];
  const float* W2  = (const float*)d_in[7];
  const float* a2s = (const float*)d_in[8];
  const float* a2d = (const float*)d_in[9];
  const float* b2  = (const float*)d_in[10];
  const float* Wg  = (const float*)d_in[11];
  const float* bg  = (const float*)d_in[12];
  float* out = (float*)d_out;

  char* base = (char*)d_ws;
  size_t off = 0;
  auto alloc = [&](size_t bytes) -> char* {
    char* p = base + off;
    off = (off + bytes + 255) & ~(size_t)255;
    return p;
  };
  float*    p1    = (float*)   alloc(16 * 4);
  float*    q2    = (float*)   alloc(2048 * 4);
  f16*      w2t   = (f16*)     alloc((size_t)2048 * 256 * 2);
  float*    as1   = (float*)   alloc((size_t)NN * 4 * 4);
  float*    ad1   = (float*)   alloc((size_t)NN * 4 * 4);
  unsigned* m1    = (unsigned*)alloc((size_t)NN * 4 * 4);
  float*    den1  = (float*)   alloc((size_t)NN * 4 * 4);
  float*    ev1   = (float*)   alloc((size_t)E2 * 4 * 4);
  float*    agg1  = (float*)   alloc((size_t)NN * HID * 4);   // becomes h1
  f16*      h1f16 = (f16*)     alloc((size_t)NN * HID * 2);
  float*    as2   = (float*)   alloc((size_t)NN * 4 * 4);
  float*    ad2   = (float*)   alloc((size_t)NN * 4 * 4);
  unsigned* m2    = (unsigned*)alloc((size_t)NN * 4 * 4);
  float*    den2  = (float*)   alloc((size_t)NN * 4 * 4);
  float*    ev2   = (float*)   alloc((size_t)E2 * 4 * 4);
  f16*      xh2   = (f16*)     alloc((size_t)NN * 2048 * 2);
  float*    agg2  = (float*)   alloc((size_t)NN * OUTC * 4);  // becomes h2
  float*    gateb = (float*)   alloc((size_t)NN * 4);
  (void)ws_size; (void)in_sizes; (void)n_in; (void)out_size;

  hipMemsetAsync(m1,   0, (size_t)NN * 4 * 4, stream);
  hipMemsetAsync(den1, 0, (size_t)NN * 4 * 4, stream);
  hipMemsetAsync(agg1, 0, (size_t)NN * HID * 4, stream);
  hipMemsetAsync(m2,   0, (size_t)NN * 4 * 4, stream);
  hipMemsetAsync(den2, 0, (size_t)NN * 4 * 4, stream);
  hipMemsetAsync(agg2, 0, (size_t)NN * OUTC * 4, stream);

  prep_p1 <<<1, 16, 0, stream>>>(W1, a1s, a1d, p1);
  prep_q2 <<<8, 256, 0, stream>>>(W2, a2s, a2d, q2);
  prep_w2t<<<2048, 256, 0, stream>>>(W2, w2t);

  node1_alpha<<<(NN + 255) / 256, 256, 0, stream>>>(x, p1, as1, ad1);
  edge_logits_max<<<(E2 + 255) / 256, 256, 0, stream>>>(ei, as1, ad1, ev1, m1);
  edge_exp_sum   <<<(E2 + 255) / 256, 256, 0, stream>>>(ei, ev1, m1, den1);
  edge1_agg      <<<E2 / 16, 256, 0, stream>>>(ei, x, W1, ev1, den1, agg1);
  post1<<<(NN * HID) / 256, 256, 0, stream>>>(agg1, b1, h1f16);

  node2_alpha<<<(NN * 8) / 256, 256, 0, stream>>>(agg1, q2, as2, ad2);
  gemm2_wmma<<<dim3(NN / 16, 2048 / 256), 128, 0, stream>>>(h1f16, w2t, xh2);

  edge_logits_max<<<(E2 + 255) / 256, 256, 0, stream>>>(ei, as2, ad2, ev2, m2);
  edge_exp_sum   <<<(E2 + 255) / 256, 256, 0, stream>>>(ei, ev2, m2, den2);
  edge2_agg      <<<E2 / 8, 256, 0, stream>>>(ei, xh2, ev2, den2, agg2);
  post2<<<(NN * OUTC) / 256, 256, 0, stream>>>(agg2, b2);

  gate_k<<<(NN + 255) / 256, 256, 0, stream>>>(agg2, Wg, bg, gateb);
  pool_k<<<BGR, 256, 0, stream>>>(agg2, gateb, out);
}
